// SimplePointNet2_V2_68410239091218
// MI455X (gfx1250) — compile-verified, matrix operand order checked by reading, not run on hardware
//
#include <hip/hip_runtime.h>
#include <hip/hip_bf16.h>
#include <stdint.h>

typedef __bf16 bf16;
typedef __attribute__((ext_vector_type(16))) __bf16 v16bf;
typedef __attribute__((ext_vector_type(8)))  float  v8f;

#define DEVINL __device__ __forceinline__

// ---------------------------------------------------------------------------
// WMMA helpers (CDNA5: D = A(16x32 bf16) x B(32x16 bf16) + C(16x16 f32))
// ---------------------------------------------------------------------------
DEVINL v8f wmma_bf16(v16bf a, v16bf b, v8f c) {
  return __builtin_amdgcn_wmma_f32_16x16x32_bf16(
      /*neg_a=*/false, a, /*neg_b=*/false, b,
      /*c_mod=*/(short)0, c, /*reuse_a=*/false, /*reuse_b=*/false);
}

// A tile from row-major bf16 matrix (row stride = `stride` elements).
// ISA 16-bit A layout (16x32): lane l: m = l%16, half = l/16;
// element e<8 -> k = half*8+e ; e>=8 -> k = 16+half*8+(e-8). Pairs are contiguous.
DEVINL v16bf load_a_tile(const bf16* A, int row0, int k0, int stride) {
  int lane = threadIdx.x & 31;
  int m    = row0 + (lane & 15);
  int half = lane >> 4;
  const bf16* base = A + (size_t)m * stride + k0 + half * 8;
  v16bf a;
  uint32_t* ap = (uint32_t*)&a;
#pragma unroll
  for (int p = 0; p < 4; ++p) {
    ap[p]     = *(const uint32_t*)(base + 2 * p);
    ap[p + 4] = *(const uint32_t*)(base + 16 + 2 * p);
  }
  return a;
}

// B tiles are pre-swizzled: tile = 512 contiguous bf16 laid out lane-major
// (lane*16 + e), so each lane loads 32 contiguous bytes.
DEVINL v16bf load_b_tile(const bf16* W, int tile) {
  const bf16* p = W + (size_t)tile * 512 + (size_t)(threadIdx.x & 31) * 16;
  v16bf b;
  ((uint4*)&b)[0] = ((const uint4*)p)[0];
  ((uint4*)&b)[1] = ((const uint4*)p)[1];
  return b;
}

// monotonic float<->uint mapping for atomicMax on floats
DEVINL unsigned fkey(float f) {
  unsigned u = __float_as_uint(f);
  return (u & 0x80000000u) ? ~u : (u | 0x80000000u);
}
DEVINL float funkey(unsigned k) {
  unsigned u = (k & 0x80000000u) ? (k & 0x7fffffffu) : ~k;
  return __uint_as_float(u);
}

// ---------------------------------------------------------------------------
// Weight conversion f32 row-major [K][N] -> swizzled bf16 WMMA-B tiles
// ---------------------------------------------------------------------------
__global__ void k_swizzle_w(const float* __restrict__ src, bf16* __restrict__ dst,
                            int K, int N, int KP, int NP) {
  int total = (KP / 32) * (NP / 16) * 512;
  for (int i = blockIdx.x * blockDim.x + threadIdx.x; i < total;
       i += gridDim.x * blockDim.x) {
    int tile = i >> 9;
    int r    = i & 511;
    int lane = r >> 4;
    int e    = r & 15;
    int tilesN = NP / 16;
    int kt = tile / tilesN, nt = tile % tilesN;
    int n    = nt * 16 + (lane & 15);
    int half = lane >> 4;
    int k    = kt * 32 + ((e < 8) ? e : e + 8) + half * 8;
    float v  = (k < K && n < N) ? src[(size_t)k * N + n] : 0.0f;
    dst[i] = (bf16)v;
  }
}

// ---------------------------------------------------------------------------
// Farthest point sampling: one persistent workgroup, positions in registers
// ---------------------------------------------------------------------------
template <int NPT>
__global__ __launch_bounds__(1024) void k_fps(const float* __restrict__ pos,
                                              int npts, int nsample,
                                              int* __restrict__ idx_out) {
  __shared__ float sv[1024];
  __shared__ int   si[1024];
  __shared__ int   s_last;
  int tid = threadIdx.x;
  float px[NPT], py[NPT], pz[NPT], mind[NPT];
#pragma unroll
  for (int i = 0; i < NPT; ++i) {
    int p = tid + i * 1024;
    px[i] = pos[p * 3 + 0];
    py[i] = pos[p * 3 + 1];
    pz[i] = pos[p * 3 + 2];
    mind[i] = 1e30f;
  }
  if (tid == 0) { idx_out[0] = 0; s_last = 0; }
  __syncthreads();
  int last = 0;
  for (int step = 1; step < nsample; ++step) {
    float lx = pos[last * 3 + 0], ly = pos[last * 3 + 1], lz = pos[last * 3 + 2];
    float bv = -1.0f;
    int   bi = 0x7fffffff;
#pragma unroll
    for (int i = 0; i < NPT; ++i) {
      float dx = px[i] - lx, dy = py[i] - ly, dz = pz[i] - lz;
      float d  = dx * dx + dy * dy + dz * dz;
      if (d < mind[i]) mind[i] = d;
      int p = tid + i * 1024;
      if (mind[i] > bv || (mind[i] == bv && p < bi)) { bv = mind[i]; bi = p; }
    }
    sv[tid] = bv; si[tid] = bi;
    __syncthreads();
    if (tid < 32) {
      float v = sv[tid * 32];
      int  ix = si[tid * 32];
      for (int j = 1; j < 32; ++j) {
        float v2 = sv[tid * 32 + j]; int i2 = si[tid * 32 + j];
        if (v2 > v || (v2 == v && i2 < ix)) { v = v2; ix = i2; }
      }
      sv[tid] = v; si[tid] = ix;
    }
    __syncthreads();
    if (tid == 0) {
      float v = sv[0]; int ix = si[0];
      for (int j = 1; j < 32; ++j)
        if (sv[j] > v || (sv[j] == v && si[j] < ix)) { v = sv[j]; ix = si[j]; }
      idx_out[step] = ix;
      s_last = ix;
    }
    __syncthreads();
    last = s_last;
  }
}

__global__ void k_gather_pos(const float* __restrict__ pos, const int* __restrict__ idx,
                             float* __restrict__ out, int n) {
  int i = blockIdx.x * blockDim.x + threadIdx.x;
  if (i < n * 3) { int r = i / 3, c = i % 3; out[i] = pos[(size_t)idx[r] * 3 + c]; }
}

// ---------------------------------------------------------------------------
// Brute-force top-64 neighbors (ascending d2) + radius-validity mask
// ---------------------------------------------------------------------------
__global__ void k_topk64(const float* __restrict__ qpos, int nq,
                         const float* __restrict__ spos, int ns, float r2,
                         int* __restrict__ nbr, unsigned long long* __restrict__ vmask) {
  int q = blockIdx.x * blockDim.x + threadIdx.x;
  if (q >= nq) return;
  float qx = qpos[q * 3], qy = qpos[q * 3 + 1], qz = qpos[q * 3 + 2];
  float bd[64];
  int   bi[64];
#pragma unroll
  for (int j = 0; j < 64; ++j) { bd[j] = 1e30f; bi[j] = 0; }
  for (int s = 0; s < ns; ++s) {
    float dx = spos[s * 3] - qx, dy = spos[s * 3 + 1] - qy, dz = spos[s * 3 + 2] - qz;
    float d  = dx * dx + dy * dy + dz * dz;
    if (d < bd[63]) {
      int j = 63;
      while (j > 0 && bd[j - 1] > d) { bd[j] = bd[j - 1]; bi[j] = bi[j - 1]; --j; }
      bd[j] = d; bi[j] = s;
    }
  }
  unsigned long long m = 0ull;
  for (int j = 0; j < 64; ++j) {
    nbr[(size_t)q * 64 + j] = bi[j];
    if (bd[j] <= r2) m |= (1ull << j);
  }
  vmask[q] = m;
}

// ---------------------------------------------------------------------------
// Fused SA-stage MLP: gather -> 3x WMMA layers in LDS -> masked max-pool
// ---------------------------------------------------------------------------
template <int KP, int NN>
DEVINL void layer_relu(const bf16* Ain, const bf16* W, const float* bias, bf16* Bout) {
  int wave = threadIdx.x >> 5;
  int lane = threadIdx.x & 31;
  const int tilesN = NN / 16;
  for (int t = wave; t < 4 * tilesN; t += 8) {
    int rt = t / tilesN, nt = t % tilesN;
    v8f acc = {};
#pragma unroll
    for (int kt = 0; kt < KP / 32; ++kt) {
      v16bf a = load_a_tile(Ain, rt * 16, kt * 32, KP);
      v16bf b = load_b_tile(W, kt * tilesN + nt);
      acc = wmma_bf16(a, b, acc);
    }
    int n   = nt * 16 + (lane & 15);
    float bv = bias[n];
    int mbase = rt * 16 + (lane >> 4) * 8;
#pragma unroll
    for (int rr = 0; rr < 8; ++rr) {
      float v = fmaxf(acc[rr] + bv, 0.0f);
      Bout[(mbase + rr) * NN + n] = (bf16)v;
    }
  }
}

template <int KP, int NN>
DEVINL void layer_maxpool(const bf16* Ain, const bf16* W, const float* bias,
                          unsigned long long vm, unsigned* sMax) {
  int wave = threadIdx.x >> 5;
  int lane = threadIdx.x & 31;
  const int tilesN = NN / 16;
  for (int t = wave; t < 4 * tilesN; t += 8) {
    int rt = t / tilesN, nt = t % tilesN;
    v8f acc = {};
#pragma unroll
    for (int kt = 0; kt < KP / 32; ++kt) {
      v16bf a = load_a_tile(Ain, rt * 16, kt * 32, KP);
      v16bf b = load_b_tile(W, kt * tilesN + nt);
      acc = wmma_bf16(a, b, acc);
    }
    int n   = nt * 16 + (lane & 15);
    float bv = bias[n];
    int mbase = rt * 16 + (lane >> 4) * 8;
#pragma unroll
    for (int rr = 0; rr < 8; ++rr) {
      int mm = mbase + rr;
      if ((vm >> mm) & 1ull) {
        float v = acc[rr] + bv;   // no ReLU on last layer
        atomicMax((unsigned int*)&sMax[n], fkey(v));
      }
    }
  }
}

template <int CF, int K0P, int H1, int H2, int CO>
__global__ __launch_bounds__(256) void k_sa_mlp(
    const float* __restrict__ xsrc, const float* __restrict__ spos,
    const float* __restrict__ qpos, const int* __restrict__ nbr,
    const unsigned long long* __restrict__ vmask,
    const bf16* __restrict__ W1, const float* __restrict__ b1,
    const bf16* __restrict__ W2, const float* __restrict__ b2,
    const bf16* __restrict__ W3, const float* __restrict__ b3,
    float* __restrict__ out) {
  __shared__ __attribute__((aligned(16))) bf16 A0[64 * K0P];
  __shared__ __attribute__((aligned(16))) bf16 Hs1[64 * H1];
  __shared__ __attribute__((aligned(16))) bf16 Hs2[64 * H2];
  __shared__ unsigned sMax[CO];
  int q   = blockIdx.x;
  int tid = threadIdx.x;
  float qx = qpos[q * 3], qy = qpos[q * 3 + 1], qz = qpos[q * 3 + 2];
  for (int c = tid; c < CO; c += 256) sMax[c] = fkey(-1e30f);
  for (int i = tid; i < 64 * K0P; i += 256) {
    int j = i / K0P, k = i - j * K0P;
    int n = nbr[(size_t)q * 64 + j];
    float v;
    if (k < CF)            v = xsrc[(size_t)n * CF + k];
    else if (k == CF)      v = spos[n * 3 + 0] - qx;
    else if (k == CF + 1)  v = spos[n * 3 + 1] - qy;
    else if (k == CF + 2)  v = spos[n * 3 + 2] - qz;
    else                   v = 0.0f;
    A0[i] = (bf16)v;
  }
  __syncthreads();
  layer_relu<K0P, H1>(A0, W1, b1, Hs1);
  __syncthreads();
  layer_relu<H1, H2>(Hs1, W2, b2, Hs2);
  __syncthreads();
  layer_maxpool<H2, CO>(Hs2, W3, b3, vmask[q], sMax);
  __syncthreads();
  for (int c = tid; c < CO; c += 256) out[(size_t)q * CO + c] = funkey(sMax[c]);
}

// ---------------------------------------------------------------------------
// Generic WMMA GEMM: out = act(A[MxKP]bf16 @ W[KPxNN]bf16 + bias)
// MODE 0: ReLU -> bf16 out [M x NN]; MODE 1: f32 out (no ReLU), cols<nout, row stride ostride
// ---------------------------------------------------------------------------
template <int KP, int NN, int MODE>
__global__ __launch_bounds__(256) void k_gemm(const bf16* __restrict__ A,
                                              const bf16* __restrict__ W,
                                              const float* __restrict__ bias,
                                              void* __restrict__ outp, int M,
                                              int nout, int ostride) {
  const int tilesN = NN / 16;
  int tiles = (M / 16) * tilesN;
  int t = blockIdx.x * 8 + (threadIdx.x >> 5);
  if (t >= tiles) return;
  int lane = threadIdx.x & 31;
  int rt = t / tilesN, nt = t % tilesN;
  v8f acc = {};
  for (int kt = 0; kt < KP / 32; ++kt) {
    v16bf a = load_a_tile(A, rt * 16, kt * 32, KP);
    v16bf b = load_b_tile(W, kt * tilesN + nt);
    acc = wmma_bf16(a, b, acc);
  }
  int n = nt * 16 + (lane & 15);
  float bv = (MODE == 1) ? ((n < nout) ? bias[n] : 0.0f) : bias[n];
  int mbase = rt * 16 + (lane >> 4) * 8;
#pragma unroll
  for (int rr = 0; rr < 8; ++rr) {
    int mm  = mbase + rr;
    float v = acc[rr] + bv;
    if (MODE == 0) {
      ((bf16*)outp)[(size_t)mm * NN + n] = (bf16)fmaxf(v, 0.0f);
    } else {
      if (n < nout) ((float*)outp)[(size_t)mm * ostride + n] = v;
    }
  }
}

// ---------------------------------------------------------------------------
// Misc small kernels
// ---------------------------------------------------------------------------
__global__ void k_colmax(const float* __restrict__ A, float* __restrict__ out,
                         int M, int N) {
  int c = blockIdx.x * blockDim.x + threadIdx.x;
  if (c >= N) return;
  float m = -1e30f;
  for (int r = 0; r < M; ++r) m = fmaxf(m, A[(size_t)r * N + c]);
  out[c] = m;
}

__global__ void k_knn3(const float* __restrict__ xp, int ns,
                       const float* __restrict__ yp, int nt,
                       int* __restrict__ idx3, float* __restrict__ w3) {
  int t = blockIdx.x * blockDim.x + threadIdx.x;
  if (t >= nt) return;
  float qx = yp[t * 3], qy = yp[t * 3 + 1], qz = yp[t * 3 + 2];
  float d0 = 1e30f, d1 = 1e30f, d2 = 1e30f;
  int   i0 = 0, i1 = 0, i2 = 0;
  for (int s = 0; s < ns; ++s) {
    float dx = xp[s * 3] - qx, dy = xp[s * 3 + 1] - qy, dz = xp[s * 3 + 2] - qz;
    float d  = dx * dx + dy * dy + dz * dz;
    if (d < d0)      { d2 = d1; i2 = i1; d1 = d0; i1 = i0; d0 = d; i0 = s; }
    else if (d < d1) { d2 = d1; i2 = i1; d1 = d;  i1 = s; }
    else if (d < d2) { d2 = d;  i2 = s; }
  }
  idx3[t * 3] = i0; idx3[t * 3 + 1] = i1; idx3[t * 3 + 2] = i2;
  w3[t * 3]     = 1.0f / (d0 + 1e-16f);
  w3[t * 3 + 1] = 1.0f / (d1 + 1e-16f);
  w3[t * 3 + 2] = 1.0f / (d2 + 1e-16f);
}

__global__ void k_build_interp_concat(const float* __restrict__ feat, int Ci,
                                      const int* __restrict__ idx3,
                                      const float* __restrict__ w3,
                                      const float* __restrict__ skip, int Cs,
                                      bf16* __restrict__ dst, int KP, int M) {
  int i = blockIdx.x * blockDim.x + threadIdx.x;
  if (i >= M * KP) return;
  int t = i / KP, k = i - t * KP;
  float v = 0.0f;
  if (k < Ci) {
    int i0 = idx3[t * 3], i1 = idx3[t * 3 + 1], i2 = idx3[t * 3 + 2];
    float w0 = w3[t * 3], w1 = w3[t * 3 + 1], w2 = w3[t * 3 + 2];
    v = (w0 * feat[(size_t)i0 * Ci + k] + w1 * feat[(size_t)i1 * Ci + k] +
         w2 * feat[(size_t)i2 * Ci + k]) / (w0 + w1 + w2);
  } else if (k < Ci + Cs) {
    v = skip[(size_t)t * Cs + (k - Ci)];
  }
  dst[i] = (bf16)v;
}

__global__ void k_build_fp4(const float* __restrict__ g, const float* __restrict__ s3x,
                            bf16* __restrict__ dst) {
  int i = blockIdx.x * blockDim.x + threadIdx.x;
  if (i >= 256 * 2304) return;
  int t = i / 2304, k = i % 2304;
  float v = (k < 2048) ? g[k] : s3x[(size_t)t * 256 + (k - 2048)];
  dst[i] = (bf16)v;
}

__global__ void k_build_sa4(const float* __restrict__ s3x, const float* __restrict__ s3p,
                            bf16* __restrict__ dst) {
  int i = blockIdx.x * blockDim.x + threadIdx.x;
  if (i >= 256 * 288) return;
  int t = i / 288, k = i % 288;
  float v = 0.0f;
  if (k < 256)      v = s3x[(size_t)t * 256 + k];
  else if (k < 259) v = s3p[t * 3 + (k - 256)];
  dst[i] = (bf16)v;
}

__global__ void k_cvt_bf16(const float* __restrict__ src, bf16* __restrict__ dst, int n) {
  int i = blockIdx.x * blockDim.x + threadIdx.x;
  if (i < n) dst[i] = (bf16)src[i];
}

// ---------------------------------------------------------------------------
// Host orchestration
// ---------------------------------------------------------------------------
struct LayerW { int K, N, KP, NP; };
// LW order: sa1(3), sa2(3), sa3(3), sa4(3), fp4(2), fp3(2), fp2(2), fp1(3), head(2)
static const LayerW LW[23] = {
  {9,32,32,32},{32,32,32,32},{32,64,32,64},
  {67,64,96,64},{64,64,64,64},{64,128,64,128},
  {131,128,160,128},{128,128,128,128},{128,256,128,256},
  {259,512,288,512},{512,1024,512,1024},{1024,2048,1024,2048},
  {2304,1024,2304,1024},{1024,512,1024,512},
  {640,256,640,256},{256,128,256,128},
  {192,128,192,128},{128,128,128,128},
  {134,64,160,64},{64,64,64,64},{64,64,64,64},
  {64,32,64,32},{32,13,32,16}
};

#define CEILDIV(a, b) (((a) + (b) - 1) / (b))
#define GEMM(KPv, NNv, MODEv, Ap, Wp, Bp, Op, Mv, NOUTv, OSTRv)                    \
  k_gemm<KPv, NNv, MODEv><<<CEILDIV(((Mv) / 16) * ((NNv) / 16), 8), 256, 0,        \
                            stream>>>((Ap), (Wp), (Bp), (Op), (Mv), (NOUTv), (OSTRv))

extern "C" void kernel_launch(void* const* d_in, const int* in_sizes, int n_in,
                              void* d_out, int out_size, void* d_ws, size_t ws_size,
                              hipStream_t stream) {
  const int N1 = 16384, NQ1 = 4096, NQ2 = 1024, NQ3 = 256;

  // --- locate inputs (handle insertion-order vs key-sorted pytree flattening) ---
  const float *x, *pos;
  int pbase;
  bool top_ins = (in_sizes[0] == N1 * 6);
  if (top_ins) { // x, pos, batch, params...
    x = (const float*)d_in[0];
    pos = (const float*)d_in[1];
    pbase = 3;
  } else {       // batch, params..., pos, x
    x = (const float*)d_in[48];
    pos = (const float*)d_in[47];
    pbase = 1;
  }
  static const int gsz[9]          = {3, 3, 3, 3, 2, 2, 2, 3, 2};
  static const int lwStartIns[9]   = {0, 3, 6, 9, 12, 14, 16, 18, 21};
  static const int lwStartSort[9]  = {11, 14, 17, 20, 7, 5, 3, 0, 9};
  bool grp_ins = (in_sizes[pbase] == 9 * 32);
  const float *Wsrc[23], *Bsrc[23];
  {
    int l = 0;
    for (int g = 0; g < 9; ++g) {
      int base = grp_ins ? lwStartIns[g] : lwStartSort[g];
      for (int o = 0; o < gsz[g]; ++o, ++l) {
        int slot = base + o;
        Wsrc[l] = (const float*)d_in[pbase + 2 * slot];
        Bsrc[l] = (const float*)d_in[pbase + 2 * slot + 1];
      }
    }
  }

  // --- workspace bump allocator ---
  char* wsb = (char*)d_ws;
  size_t off = 0;
  auto alloc = [&](size_t bytes) -> void* {
    void* p = wsb + off;
    off = (off + bytes + 255) & ~(size_t)255;
    return p;
  };

  // --- convert all weights to swizzled bf16 WMMA-B layout ---
  bf16* Wb[23];
  for (int l = 0; l < 23; ++l) {
    size_t elems = (size_t)(LW[l].KP / 32) * (LW[l].NP / 16) * 512;
    Wb[l] = (bf16*)alloc(elems * sizeof(bf16));
    int blocks = (int)CEILDIV(elems, (size_t)256);
    if (blocks > 8192) blocks = 8192;
    k_swizzle_w<<<blocks, 256, 0, stream>>>(Wsrc[l], Wb[l], LW[l].K, LW[l].N,
                                            LW[l].KP, LW[l].NP);
  }

  // --- pipeline buffers ---
  int* idx1 = (int*)alloc((size_t)NQ1 * 4);
  int* idx2 = (int*)alloc((size_t)NQ2 * 4);
  int* idx3q = (int*)alloc((size_t)NQ3 * 4);
  float* pos1 = (float*)alloc((size_t)NQ1 * 3 * 4);
  float* pos2 = (float*)alloc((size_t)NQ2 * 3 * 4);
  float* pos3 = (float*)alloc((size_t)NQ3 * 3 * 4);
  int* nbr1 = (int*)alloc((size_t)NQ1 * 64 * 4);
  int* nbr2 = (int*)alloc((size_t)NQ2 * 64 * 4);
  int* nbr3 = (int*)alloc((size_t)NQ3 * 64 * 4);
  unsigned long long* msk1 = (unsigned long long*)alloc((size_t)NQ1 * 8);
  unsigned long long* msk2 = (unsigned long long*)alloc((size_t)NQ2 * 8);
  unsigned long long* msk3 = (unsigned long long*)alloc((size_t)NQ3 * 8);
  float* s1x = (float*)alloc((size_t)NQ1 * 64 * 4);
  float* s2x = (float*)alloc((size_t)NQ2 * 128 * 4);
  float* s3x = (float*)alloc((size_t)NQ3 * 256 * 4);

  // ============ SA1 ============
  k_fps<16><<<1, 1024, 0, stream>>>(pos, N1, NQ1, idx1);
  k_gather_pos<<<CEILDIV(NQ1 * 3, 256), 256, 0, stream>>>(pos, idx1, pos1, NQ1);
  k_topk64<<<CEILDIV(NQ1, 64), 64, 0, stream>>>(pos1, NQ1, pos, N1, 0.04f, nbr1, msk1);
  k_sa_mlp<6, 32, 32, 32, 64><<<NQ1, 256, 0, stream>>>(
      x, pos, pos1, nbr1, msk1, Wb[0], Bsrc[0], Wb[1], Bsrc[1], Wb[2], Bsrc[2], s1x);

  // ============ SA2 ============
  k_fps<4><<<1, 1024, 0, stream>>>(pos1, NQ1, NQ2, idx2);
  k_gather_pos<<<CEILDIV(NQ2 * 3, 256), 256, 0, stream>>>(pos1, idx2, pos2, NQ2);
  k_topk64<<<CEILDIV(NQ2, 64), 64, 0, stream>>>(pos2, NQ2, pos1, NQ1, 0.16f, nbr2, msk2);
  k_sa_mlp<64, 96, 64, 64, 128><<<NQ2, 256, 0, stream>>>(
      s1x, pos1, pos2, nbr2, msk2, Wb[3], Bsrc[3], Wb[4], Bsrc[4], Wb[5], Bsrc[5], s2x);

  // ============ SA3 ============
  k_fps<1><<<1, 1024, 0, stream>>>(pos2, NQ2, NQ3, idx3q);
  k_gather_pos<<<CEILDIV(NQ3 * 3, 256), 256, 0, stream>>>(pos2, idx3q, pos3, NQ3);
  k_topk64<<<CEILDIV(NQ3, 64), 64, 0, stream>>>(pos3, NQ3, pos2, NQ2, 0.64f, nbr3, msk3);
  k_sa_mlp<128, 160, 128, 128, 256><<<NQ3, 256, 0, stream>>>(
      s2x, pos2, pos3, nbr3, msk3, Wb[6], Bsrc[6], Wb[7], Bsrc[7], Wb[8], Bsrc[8], s3x);

  // ============ SA4 (global MLP + max) ============
  bf16* A4  = (bf16*)alloc((size_t)256 * 288 * 2);
  bf16* h41 = (bf16*)alloc((size_t)256 * 512 * 2);
  bf16* h42 = (bf16*)alloc((size_t)256 * 1024 * 2);
  float* o4 = (float*)alloc((size_t)256 * 2048 * 4);
  float* g  = (float*)alloc((size_t)2048 * 4);
  k_build_sa4<<<CEILDIV(256 * 288, 256), 256, 0, stream>>>(s3x, pos3, A4);
  GEMM(288, 512, 0, A4, Wb[9], Bsrc[9], h41, 256, 512, 0);
  GEMM(512, 1024, 0, h41, Wb[10], Bsrc[10], h42, 256, 1024, 0);
  GEMM(1024, 2048, 1, h42, Wb[11], Bsrc[11], o4, 256, 2048, 2048);
  k_colmax<<<8, 256, 0, stream>>>(o4, g, 256, 2048);

  // ============ FP4 ============
  bf16* Af4 = (bf16*)alloc((size_t)256 * 2304 * 2);
  bf16* hf4 = (bf16*)alloc((size_t)256 * 1024 * 2);
  float* f4 = (float*)alloc((size_t)256 * 512 * 4);
  k_build_fp4<<<CEILDIV(256 * 2304, 256), 256, 0, stream>>>(g, s3x, Af4);
  GEMM(2304, 1024, 0, Af4, Wb[12], Bsrc[12], hf4, 256, 1024, 0);
  GEMM(1024, 512, 1, hf4, Wb[13], Bsrc[13], f4, 256, 512, 512);

  // ============ FP3 ============
  int* i3a = (int*)alloc((size_t)NQ2 * 3 * 4);
  float* w3a = (float*)alloc((size_t)NQ2 * 3 * 4);
  bf16* Af3 = (bf16*)alloc((size_t)NQ2 * 640 * 2);
  bf16* hf3 = (bf16*)alloc((size_t)NQ2 * 256 * 2);
  float* f3 = (float*)alloc((size_t)NQ2 * 128 * 4);
  k_knn3<<<CEILDIV(NQ2, 64), 64, 0, stream>>>(pos3, NQ3, pos2, NQ2, i3a, w3a);
  k_build_interp_concat<<<CEILDIV(NQ2 * 640, 256), 256, 0, stream>>>(
      f4, 512, i3a, w3a, s2x, 128, Af3, 640, NQ2);
  GEMM(640, 256, 0, Af3, Wb[14], Bsrc[14], hf3, NQ2, 256, 0);
  GEMM(256, 128, 1, hf3, Wb[15], Bsrc[15], f3, NQ2, 128, 128);

  // ============ FP2 ============
  int* i2a = (int*)alloc((size_t)NQ1 * 3 * 4);
  float* w2a = (float*)alloc((size_t)NQ1 * 3 * 4);
  bf16* Af2 = (bf16*)alloc((size_t)NQ1 * 192 * 2);
  bf16* hf2 = (bf16*)alloc((size_t)NQ1 * 128 * 2);
  float* f2 = (float*)alloc((size_t)NQ1 * 128 * 4);
  k_knn3<<<CEILDIV(NQ1, 64), 64, 0, stream>>>(pos2, NQ2, pos1, NQ1, i2a, w2a);
  k_build_interp_concat<<<CEILDIV(NQ1 * 192, 256), 256, 0, stream>>>(
      f3, 128, i2a, w2a, s1x, 64, Af2, 192, NQ1);
  GEMM(192, 128, 0, Af2, Wb[16], Bsrc[16], hf2, NQ1, 128, 0);
  GEMM(128, 128, 1, hf2, Wb[17], Bsrc[17], f2, NQ1, 128, 128);

  // ============ FP1 ============
  int* i1a = (int*)alloc((size_t)N1 * 3 * 4);
  float* w1a = (float*)alloc((size_t)N1 * 3 * 4);
  bf16* Af1 = (bf16*)alloc((size_t)N1 * 160 * 2);
  bf16* hA  = (bf16*)alloc((size_t)N1 * 64 * 2);
  bf16* hB  = (bf16*)alloc((size_t)N1 * 64 * 2);
  float* f1 = (float*)alloc((size_t)N1 * 64 * 4);
  k_knn3<<<CEILDIV(N1, 64), 64, 0, stream>>>(pos1, NQ1, pos, N1, i1a, w1a);
  k_build_interp_concat<<<CEILDIV(N1 * 160, 256), 256, 0, stream>>>(
      f2, 128, i1a, w1a, x, 6, Af1, 160, N1);
  GEMM(160, 64, 0, Af1, Wb[18], Bsrc[18], hA, N1, 64, 0);
  GEMM(64, 64, 0, hA, Wb[19], Bsrc[19], hB, N1, 64, 0);
  GEMM(64, 64, 1, hB, Wb[20], Bsrc[20], f1, N1, 64, 64);

  // ============ HEAD ============
  bf16* Ah = (bf16*)alloc((size_t)N1 * 64 * 2);
  bf16* hh = (bf16*)alloc((size_t)N1 * 32 * 2);
  k_cvt_bf16<<<CEILDIV(N1 * 64, 256), 256, 0, stream>>>(f1, Ah, N1 * 64);
  GEMM(64, 32, 0, Ah, Wb[21], Bsrc[21], hh, N1, 32, 0);
  GEMM(32, 16, 1, hh, Wb[22], Bsrc[22], (float*)d_out, N1, 13, 13);

  (void)n_in; (void)out_size; (void)ws_size;
}
